// EncoderGRUODE_26886495272993
// MI455X (gfx1250) — matrix operand order repaired and checked by reading, not compiled, not run on hardware
//
#include <hip/hip_runtime.h>
#include <hip/hip_bf16.h>

typedef unsigned short ushort_t;
typedef __attribute__((ext_vector_type(16))) __bf16 v16bf;
typedef __attribute__((ext_vector_type(8)))  __bf16 v8bf;
typedef __attribute__((ext_vector_type(8)))  float  v8f;

#define B_SZ   256
#define T_ENC_ 64
#define T_DEC_ 32
#define X_DIM_ 128
#define F_DIM_ 64
#define D_IN_  192
#define H_DIM  1024
#define HN_DIM 2048
#define H3     3072
#define T_ALL  96

// ---------------- helpers ----------------
__device__ __forceinline__ ushort_t f2bf_rne(float f) {
  union { float f; unsigned u; } x; x.f = f;
  unsigned u = x.u;
  unsigned r = u + 0x7FFFu + ((u >> 16) & 1u);
  return (ushort_t)(r >> 16);
}
__device__ __forceinline__ float sigmoidf_(float x) { return 1.0f / (1.0f + expf(-x)); }

// ---------------- weight tiling: W (N x K) fp32 -> bf16 B-fragments ----------------
// Per 16x32 (NxK) tile: lane L holds column n = nt*16 + L%16; 16 K values:
// lanes 0-15 -> K = kt*32 + {0..7, 16..23}, lanes 16-31 -> K = kt*32 + {8..15, 24..31}.
// Flat: Wt[((nt*KT32 + kt)*32 + lane)*16 + j]
__global__ void tile_weights(const float* __restrict__ W, ushort_t* __restrict__ Wt, int K) {
  const int nt = blockIdx.x;
  const int kt = blockIdx.y;
  const int lane = threadIdx.x;
  const int kt32 = gridDim.y;
  const int n  = nt * 16 + (lane & 15);
  const int kb = kt * 32 + ((lane & 16) ? 8 : 0);
  ushort_t* dst = Wt + ((size_t)(nt * kt32 + kt) * 32 + lane) * 16;
  const float* src = W + (size_t)n * K + kb;
#pragma unroll
  for (int j = 0; j < 16; ++j) dst[j] = f2bf_rne(src[j < 8 ? j : j + 8]);
}

// ---------------- WMMA GEMM: out = epilogue(A @ W^T + bias) ----------------
// A: M=256 x K bf16 row-major (pre-converted by producers). W tiled as above.
// Each wave: 16 rows x 128 cols (8 accumulators), K-loop in steps of 32.
enum { MODE_TANH_BF16 = 0, MODE_PLAIN = 1, MODE_ODE_FIRST = 2, MODE_ODE_MID = 3, MODE_ODE_LAST = 4 };

template <int MODE>
__global__ __launch_bounds__(128) void gemm_wmma(
    const ushort_t* __restrict__ A, int lda,
    const ushort_t* __restrict__ Wt, const float* __restrict__ bias,
    int N, int K,
    float* outf, ushort_t* outb, int ldo,
    const float* hbase, float* accb, float* tmpo, ushort_t* shadowb,
    float ctmp, float dt6)
{
  const int lane  = threadIdx.x & 31;
  const int wave  = blockIdx.x * (blockDim.x >> 5) + (threadIdx.x >> 5);
  const int n128  = N >> 7;
  const int mt    = wave / n128;
  const int ng    = wave - mt * n128;
  const int m0    = mt << 4;
  const int kt32  = K >> 5;

  v8f acc[8];
#pragma unroll
  for (int t = 0; t < 8; ++t) acc[t] = (v8f){};

  // A-fragment addressing (ISA 16-bit A layout):
  // lanes 0-15: row m0+lane, K = k0+{0..7} then k0+{16..23}
  // lanes 16-31: row m0+lane-16, K = k0+{8..15} then k0+{24..31}
  const int arow  = m0 + (lane & 15);
  const int akoff = (lane & 16) ? 8 : 0;
  const ushort_t* Abase = A + (size_t)arow * lda + akoff;

  const size_t ntile_stride = (size_t)kt32 * 512;  // ushorts between n-tiles
  const ushort_t* Bbase = Wt + ((size_t)(ng * 8) * kt32 * 32 + lane) * 16;

  for (int kt = 0; kt < kt32; ++kt) {
    const ushort_t* pa = Abase + kt * 32;
    v8bf alo = *(const v8bf*)(pa);
    v8bf ahi = *(const v8bf*)(pa + 16);
    v16bf a = __builtin_shufflevector(alo, ahi, 0, 1, 2, 3, 4, 5, 6, 7,
                                      8, 9, 10, 11, 12, 13, 14, 15);
    const ushort_t* bp = Bbase + (size_t)kt * 512;
#pragma unroll
    for (int t = 0; t < 8; ++t) {
      v16bf b = *(const v16bf*)(bp + (size_t)t * ntile_stride);
      acc[t] = __builtin_amdgcn_wmma_f32_16x16x32_bf16(false, a, false, b, (short)0,
                                                       acc[t], false, false);
    }
  }

  // Epilogue. C/D layout: element i of v8f -> row m0 + i + (lane<16?0:8), col = n0 + lane%16.
  const int col0 = (ng << 7) + (lane & 15);
  const int row0 = m0 + ((lane & 16) ? 8 : 0);
#pragma unroll
  for (int j = 0; j < 8; ++j) {
    const int col = col0 + j * 16;
    const float bcol = bias[col];
#pragma unroll
    for (int i = 0; i < 8; ++i) {
      const int row = row0 + i;
      const float v = acc[j][i] + bcol;
      const size_t idx = (size_t)row * ldo + col;
      if (MODE == MODE_TANH_BF16) {
        outb[idx] = f2bf_rne(tanhf(v));
      } else if (MODE == MODE_PLAIN) {
        outf[idx] = v;
      } else if (MODE == MODE_ODE_FIRST) {        // k1: acc = k; tmp = h + c*k
        accb[idx] = v;
        const float tv = hbase[idx] + ctmp * v;
        tmpo[idx] = tv;
        shadowb[idx] = f2bf_rne(tv);
      } else if (MODE == MODE_ODE_MID) {          // k2/k3: acc += 2k; tmp = h + c*k
        accb[idx] += 2.0f * v;
        const float tv = hbase[idx] + ctmp * v;
        tmpo[idx] = tv;
        shadowb[idx] = f2bf_rne(tv);
      } else {                                    // k4: h = h + dt/6*(acc + k)
        const float hv = hbase[idx] + dt6 * (accb[idx] + v);
        outf[idx] = hv;
        shadowb[idx] = f2bf_rne(hv);
      }
    }
  }
}

// ---------------- elementwise kernels ----------------
__global__ void prep_mask(const int* __restrict__ mask, float* maskf, float* rfill) {
  int b = blockIdx.x * blockDim.x + threadIdx.x;
  if (b >= B_SZ) return;
  int cum = 0;
  for (int t = 0; t < T_ENC_; ++t) {
    int m = mask[b * T_ENC_ + t];
    maskf[b * T_ENC_ + t] = (float)m;
    rfill[b * T_ENC_ + t] = (cum > 0) ? 1.0f : 0.0f;
    cum += m;
  }
}

__global__ void fzero(float* p, int n) {
  int i = blockIdx.x * blockDim.x + threadIdx.x;
  if (i < n) p[i] = 0.0f;
}
__global__ void fzero16(ushort_t* p, int n) {
  int i = blockIdx.x * blockDim.x + threadIdx.x;
  if (i < n) p[i] = 0;
}
__global__ void fcopy(float* dst, const float* __restrict__ src, int n) {
  int i = blockIdx.x * blockDim.x + threadIdx.x;
  if (i < n) dst[i] = src[i];
}
// copy f32 -> (f32 master, bf16 shadow)
__global__ void fcopy_sh(float* dstf, ushort_t* dstb, const float* __restrict__ src, int n) {
  int i = blockIdx.x * blockDim.x + threadIdx.x;
  if (i < n) {
    float v = src[i];
    dstf[i] = v;
    dstb[i] = f2bf_rne(v);
  }
}

__global__ void ode_gate(float* hode, ushort_t* hodeb, const float* __restrict__ h,
                         const float* __restrict__ rfill, int t) {
  int i = blockIdx.x * blockDim.x + threadIdx.x;
  if (i >= B_SZ * H_DIM) return;
  int b = i >> 10;
  float r = rfill[b * T_ENC_ + t];
  float v = r * hode[i] + (1.0f - r) * h[i];
  hode[i] = v;
  hodeb[i] = f2bf_rne(v);
}

__global__ void pack_enc(ushort_t* ebufb, const float* __restrict__ x,
                         const float* __restrict__ feat, int t) {
  int i = blockIdx.x * blockDim.x + threadIdx.x;
  if (i >= B_SZ * D_IN_) return;
  int b = i / D_IN_;
  int k = i - b * D_IN_;
  float v = (k < X_DIM_)
      ? x[((size_t)b * T_ENC_ + t) * X_DIM_ + k]
      : feat[((size_t)b * T_ALL + t) * F_DIM_ + (k - X_DIM_)];
  ebufb[i] = f2bf_rne(v);
}

__global__ void pack_dec(ushort_t* ebufb, const float* __restrict__ cx,
                         const float* __restrict__ feat, int t) {
  int i = blockIdx.x * blockDim.x + threadIdx.x;
  if (i >= B_SZ * D_IN_) return;
  int b = i / D_IN_;
  int k = i - b * D_IN_;
  float v = (k < X_DIM_)
      ? cx[b * X_DIM_ + k]
      : feat[((size_t)b * T_ALL + (T_ENC_ + t)) * F_DIM_ + (k - X_DIM_)];
  ebufb[i] = f2bf_rne(v);
}

__global__ void gru_gates(float* h, ushort_t* hb,
                          const float* __restrict__ gx, const float* __restrict__ gh,
                          const float* __restrict__ hode, const float* __restrict__ maskf,
                          int t, int use_mask) {
  int i = blockIdx.x * blockDim.x + threadIdx.x;
  if (i >= B_SZ * H_DIM) return;
  int b = i >> 10;
  int j = i & (H_DIM - 1);
  size_t o = (size_t)b * H3;
  float r = sigmoidf_(gx[o + j] + gh[o + j]);
  float z = sigmoidf_(gx[o + H_DIM + j] + gh[o + H_DIM + j]);
  float n = tanhf(gx[o + 2 * H_DIM + j] + r * gh[o + 2 * H_DIM + j]);
  float hn = (1.0f - z) * n + z * hode[i];
  float m = use_mask ? maskf[b * T_ENC_ + t] : 1.0f;
  float hv = m * hn + (1.0f - m) * hode[i];
  h[i] = hv;
  hb[i] = f2bf_rne(hv);
}

__global__ void scatter_enc(float* out, const float* __restrict__ pbuf, int t) {
  int i = blockIdx.x * blockDim.x + threadIdx.x;
  if (i >= B_SZ * X_DIM_) return;
  int b = i >> 7;
  int c = i & 127;
  float v = pbuf[i];
  size_t o0 = (((size_t)b) * (T_ENC_ - 1) + t) * X_DIM_ + c;
  size_t o1 = (((size_t)B_SZ + b) * (T_ENC_ - 1) + t) * X_DIM_ + c;
  out[o0] = v;
  out[o1] = v;
}

__global__ void scatter_dec(float* out, float* cx, const float* __restrict__ pbuf, int t) {
  int i = blockIdx.x * blockDim.x + threadIdx.x;
  if (i >= B_SZ * X_DIM_) return;
  int b = i >> 7;
  int c = i & 127;
  float v = pbuf[i];
  const size_t OFF = (size_t)2 * B_SZ * (T_ENC_ - 1) * X_DIM_;
  out[OFF + (((size_t)b) * T_DEC_ + t) * X_DIM_ + c] = v;
  out[OFF + (((size_t)B_SZ + b) * T_DEC_ + t) * X_DIM_ + c] = v;
  cx[i] = v;
}

// ---------------- host orchestration ----------------
extern "C" void kernel_launch(void* const* d_in, const int* in_sizes, int n_in,
                              void* d_out, int out_size, void* d_ws, size_t ws_size,
                              hipStream_t stream) {
  const float* x     = (const float*)d_in[0];
  const float* feat  = (const float*)d_in[1];
  const int*   mask  = (const int*)d_in[2];
  const float* w_ih  = (const float*)d_in[3];
  const float* w_hh  = (const float*)d_in[4];
  const float* b_ih  = (const float*)d_in[5];
  const float* b_hh  = (const float*)d_in[6];
  const float* w1    = (const float*)d_in[7];
  const float* b1    = (const float*)d_in[8];
  const float* w2    = (const float*)d_in[9];
  const float* b2    = (const float*)d_in[10];
  const float* w_out = (const float*)d_in[11];
  const float* b_out = (const float*)d_in[12];
  float* out = (float*)d_out;

  // workspace carve (256B aligned)
  unsigned char* p = (unsigned char*)d_ws;
  auto alloc = [&](size_t bytes) -> void* {
    void* r = (void*)p;
    p += (bytes + 255) & ~(size_t)255;
    return r;
  };
  ushort_t* wihb   = (ushort_t*)alloc(sizeof(ushort_t) * (size_t)H3 * D_IN_);
  ushort_t* whhb   = (ushort_t*)alloc(sizeof(ushort_t) * (size_t)H3 * H_DIM);
  ushort_t* w1b    = (ushort_t*)alloc(sizeof(ushort_t) * (size_t)HN_DIM * H_DIM);
  ushort_t* w2b    = (ushort_t*)alloc(sizeof(ushort_t) * (size_t)H_DIM * HN_DIM);
  ushort_t* woutb  = (ushort_t*)alloc(sizeof(ushort_t) * (size_t)X_DIM_ * H_DIM);
  ushort_t* t1b    = (ushort_t*)alloc(sizeof(ushort_t) * (size_t)B_SZ * HN_DIM);
  ushort_t* hode_b = (ushort_t*)alloc(sizeof(ushort_t) * (size_t)B_SZ * H_DIM);
  ushort_t* tmp_b  = (ushort_t*)alloc(sizeof(ushort_t) * (size_t)B_SZ * H_DIM);
  ushort_t* h_b    = (ushort_t*)alloc(sizeof(ushort_t) * (size_t)B_SZ * H_DIM);
  ushort_t* ebuf_b = (ushort_t*)alloc(sizeof(ushort_t) * (size_t)B_SZ * D_IN_);
  float* h     = (float*)alloc(4ull * B_SZ * H_DIM);
  float* hode  = (float*)alloc(4ull * B_SZ * H_DIM);
  float* tmp   = (float*)alloc(4ull * B_SZ * H_DIM);
  float* accb  = (float*)alloc(4ull * B_SZ * H_DIM);
  float* gx    = (float*)alloc(4ull * B_SZ * H3);
  float* gh    = (float*)alloc(4ull * B_SZ * H3);
  float* pbuf  = (float*)alloc(4ull * B_SZ * X_DIM_);
  float* cx    = (float*)alloc(4ull * B_SZ * X_DIM_);
  float* maskf = (float*)alloc(4ull * B_SZ * T_ENC_);
  float* rfill = (float*)alloc(4ull * B_SZ * T_ENC_);

  // --- weight conversion to bf16 WMMA tiles (once per call) ---
  tile_weights<<<dim3(H3 / 16, D_IN_ / 32), 32, 0, stream>>>(w_ih, wihb, D_IN_);
  tile_weights<<<dim3(H3 / 16, H_DIM / 32), 32, 0, stream>>>(w_hh, whhb, H_DIM);
  tile_weights<<<dim3(HN_DIM / 16, H_DIM / 32), 32, 0, stream>>>(w1, w1b, H_DIM);
  tile_weights<<<dim3(H_DIM / 16, HN_DIM / 32), 32, 0, stream>>>(w2, w2b, HN_DIM);
  tile_weights<<<dim3(X_DIM_ / 16, H_DIM / 32), 32, 0, stream>>>(w_out, woutb, H_DIM);

  prep_mask<<<1, 256, 0, stream>>>(mask, maskf, rfill);

  const int NH = B_SZ * H_DIM;          // 262144
  const int GH_BLK = NH / 256;          // 1024
  const int NE = B_SZ * D_IN_;          // 49152
  const int NP = B_SZ * X_DIM_;         // 32768

  // --- GEMM launchers (grid = N/32 blocks of 128 threads = N/8 waves) ---
  auto gemm_tanh1 = [&](const ushort_t* Ain) {
    gemm_wmma<MODE_TANH_BF16><<<HN_DIM / 32, 128, 0, stream>>>(
        Ain, H_DIM, w1b, b1, HN_DIM, H_DIM, nullptr, t1b, HN_DIM,
        nullptr, nullptr, nullptr, nullptr, 0.0f, 0.0f);
  };
  auto ode_first = [&]() {
    gemm_wmma<MODE_ODE_FIRST><<<H_DIM / 32, 128, 0, stream>>>(
        t1b, HN_DIM, w2b, b2, H_DIM, HN_DIM, nullptr, nullptr, H_DIM,
        hode, accb, tmp, tmp_b, 0.25f, 0.0f);
  };
  auto ode_mid = [&](float c) {
    gemm_wmma<MODE_ODE_MID><<<H_DIM / 32, 128, 0, stream>>>(
        t1b, HN_DIM, w2b, b2, H_DIM, HN_DIM, nullptr, nullptr, H_DIM,
        hode, accb, tmp, tmp_b, c, 0.0f);
  };
  auto ode_last = [&]() {
    gemm_wmma<MODE_ODE_LAST><<<H_DIM / 32, 128, 0, stream>>>(
        t1b, HN_DIM, w2b, b2, H_DIM, HN_DIM, hode, nullptr, H_DIM,
        hode, accb, nullptr, hode_b, 0.0f, 0.5f / 6.0f);
  };
  // RK4 with dt = 1/N_ODE_STEPS = 0.5 ; two steps, in-place on hode
  auto ode_solve_seq = [&]() {
    for (int rk = 0; rk < 2; ++rk) {
      gemm_tanh1(hode_b); ode_first();       // k1: tmp = h + 0.25*k1
      gemm_tanh1(tmp_b);  ode_mid(0.25f);    // k2
      gemm_tanh1(tmp_b);  ode_mid(0.5f);     // k3: tmp = h + 0.5*k3
      gemm_tanh1(tmp_b);  ode_last();        // k4: hode += (0.5/6)*(sum)
    }
  };
  auto gemm_plain = [&](const ushort_t* Ain, int lda, const ushort_t* Wt, const float* bias,
                        int N, int K, float* outp, int ldo) {
    gemm_wmma<MODE_PLAIN><<<N / 32, 128, 0, stream>>>(
        Ain, lda, Wt, bias, N, K, outp, nullptr, ldo,
        nullptr, nullptr, nullptr, nullptr, 0.0f, 0.0f);
  };
  auto gru_step = [&](int t, int use_mask) {
    gemm_plain(ebuf_b, D_IN_, wihb, b_ih, H3, D_IN_, gx, H3);
    gemm_plain(hode_b, H_DIM, whhb, b_hh, H3, H_DIM, gh, H3);
    gru_gates<<<GH_BLK, 256, 0, stream>>>(h, h_b, gx, gh, hode, maskf, t, use_mask);
  };
  auto project = [&]() {
    gemm_plain(h_b, H_DIM, woutb, b_out, X_DIM_, H_DIM, pbuf, X_DIM_);
  };

  // ================= encoder =================
  fzero<<<GH_BLK, 256, 0, stream>>>(h, NH);
  fzero<<<GH_BLK, 256, 0, stream>>>(hode, NH);
  fzero16<<<GH_BLK, 256, 0, stream>>>(hode_b, NH);

  // t = 0 : h_first = m * gru(e_0, 0)
  pack_enc<<<NE / 256, 256, 0, stream>>>(ebuf_b, x, feat, 0);
  gru_step(0, 1);
  project();
  scatter_enc<<<NP / 256, 256, 0, stream>>>(out, pbuf, 0);

  for (int t = 1; t < T_ENC_; ++t) {
    fcopy_sh<<<GH_BLK, 256, 0, stream>>>(hode, hode_b, h, NH);
    ode_solve_seq();
    ode_gate<<<GH_BLK, 256, 0, stream>>>(hode, hode_b, h, rfill, t);
    pack_enc<<<NE / 256, 256, 0, stream>>>(ebuf_b, x, feat, t);
    gru_step(t, 1);
    project();
    if (t < T_ENC_ - 1) {
      scatter_enc<<<NP / 256, 256, 0, stream>>>(out, pbuf, t);
    } else {
      fcopy<<<NP / 256, 256, 0, stream>>>(cx, pbuf, NP);  // last_out -> cx
    }
  }

  // ================= decoder =================
  for (int t = 0; t < T_DEC_; ++t) {
    fcopy_sh<<<GH_BLK, 256, 0, stream>>>(hode, hode_b, h, NH);
    ode_solve_seq();                                       // no rfill gating
    pack_dec<<<NE / 256, 256, 0, stream>>>(ebuf_b, cx, feat, t);
    gru_step(t, 0);                                        // no mask
    project();
    scatter_dec<<<NP / 256, 256, 0, stream>>>(out, cx, pbuf, t);
  }
}
